// EfficientWindowAttention3D_29867202576462
// MI455X (gfx1250) — matrix operand order repaired and checked by reading, not compiled, not run on hardware
//
#include <hip/hip_runtime.h>
#include <hip/hip_bf16.h>

// ---------------- types ----------------
typedef __attribute__((ext_vector_type(16))) __bf16  v16bf;
typedef __attribute__((ext_vector_type(8)))  float   v8f;
typedef __attribute__((ext_vector_type(8)))  unsigned int v8u;
typedef __attribute__((ext_vector_type(4)))  unsigned int v4u;
typedef __attribute__((ext_vector_type(8)))  int v8i;
typedef __attribute__((ext_vector_type(4)))  int v4i;

#define DIM   384
#define HEADS 12
#define HD    32
#define NTOK  343
#define NPAD  352        // 22 * 16
#define BATCH 128
#define ROWS  (BATCH * NTOK)   // 43904 = 343 * 128
#define QKVC  (3 * DIM)        // 1152

__device__ __forceinline__ unsigned short f2bf(float f) {
    unsigned int u = __float_as_uint(f);
    unsigned int r = u + 0x7FFFu + ((u >> 16) & 1u);
    return (unsigned short)(r >> 16);
}

__device__ __forceinline__ v16bf mkfrag(uint4 a, uint4 b) {
    v8u u;
    u[0] = a.x; u[1] = a.y; u[2] = a.z; u[3] = a.w;
    u[4] = b.x; u[5] = b.y; u[6] = b.z; u[7] = b.w;
    return __builtin_bit_cast(v16bf, u);
}

__device__ __forceinline__ v8f wmma_bf16(v16bf a, v16bf b, v8f c) {
    // D = A(16x32 bf16) * B(32x16 bf16) + C(16x16 f32)
    return __builtin_amdgcn_wmma_f32_16x16x32_bf16(false, a, false, b,
                                                   (short)0, c, false, false);
}

// ---------------- TDM: async 2-D bf16 tile load Global -> LDS ----------------
// D# per CDNA5 ISA 8.3/8.4: group0 = {count, lds_addr, global_addr(57b), type=2},
// group1 = {data_size=1(2B), tensor_dim0/1 (remaining extent from tile origin,
// OOB reads return 0), tile_dim0/1, tensor_dim0_stride}. Groups 2/3 + second
// descriptor slot zero-filled (2-D, single descriptor).
// Tracked by TENSORcnt; EXEC ignored (wave-level issue).
__device__ __forceinline__ void tdm_load_2d_bf16(
    const unsigned short* gptr, unsigned int lds_off,
    unsigned int dim0_rem, unsigned int dim1_rem,
    unsigned int tile0, unsigned int tile1,
    unsigned long long stride0_elems) {
    unsigned long long ga = (unsigned long long)(uintptr_t)gptr;
    v4u g0;
    g0[0] = 1u;                                          // count=1 (user mode)
    g0[1] = lds_off;                                     // LDS byte address
    g0[2] = (unsigned int)(ga & 0xFFFFFFFFu);            // global_addr[31:0]
    g0[3] = (unsigned int)((ga >> 32) & 0x01FFFFFFu)     // global_addr[56:32]
          | (2u << 30);                                  // type=2 ("image")
    v8i g1;
    g1[0] = (int)(1u << 16);                             // data_size=1 -> 2 bytes
    g1[1] = (int)((dim0_rem & 0xFFFFu) << 16);           // tensor_dim0[15:0]
    g1[2] = (int)(((dim0_rem >> 16) & 0xFFFFu)
          |        ((dim1_rem & 0xFFFFu) << 16));        // dim0[31:16] | dim1[15:0]
    g1[3] = (int)(((dim1_rem >> 16) & 0xFFFFu)
          |        ((tile0 & 0xFFFFu) << 16));           // dim1[31:16] | tile_dim0
    g1[4] = (int)(tile1 & 0xFFFFu);                      // tile_dim1 (tile_dim2=0)
    g1[5] = (int)(unsigned int)(stride0_elems & 0xFFFFFFFFull);   // dim0_stride[31:0]
    g1[6] = (int)(unsigned int)((stride0_elems >> 32) & 0xFFFFull); // dim0_stride[47:32]
    g1[7] = 0;                                           // dim1_stride unused (2-D)
    v4i z4 = {0, 0, 0, 0};
    v8i z8 = {0, 0, 0, 0, 0, 0, 0, 0};
    __builtin_amdgcn_tensor_load_to_lds(g0, g1, z4, z4, z8, 0);
}

__device__ __forceinline__ unsigned int lds_offset_of(const void* p) {
    // generic LDS address = {shared aperture, lds offset}; low 32 bits = offset
    return (unsigned int)(uintptr_t)p;
}

// ---------------- kernel: fp32 -> bf16 convert ----------------
__global__ void __launch_bounds__(256) cvt_bf16_kernel(const float* __restrict__ src,
                                                       unsigned short* __restrict__ dst,
                                                       int n) {
    int i = blockIdx.x * 256 + threadIdx.x;
    int stride = gridDim.x * 256;
    for (; i < n; i += stride) dst[i] = f2bf(src[i]);
}

// ---------------- kernel: dense padded relative-position bias ----------------
// biasM[h][r][c] for r,c in [0,352): valid -> table[rel[r,c]*12+h], pad -> -1e30
__global__ void __launch_bounds__(256) build_bias_kernel(const float* __restrict__ table,
                                                         const int* __restrict__ rel,
                                                         float* __restrict__ biasM) {
    int idx = blockIdx.x * 256 + threadIdx.x;
    const int total = HEADS * NPAD * NPAD;
    if (idx >= total) return;
    int h = idx / (NPAD * NPAD);
    int r = (idx / NPAD) % NPAD;
    int c = idx % NPAD;
    float v = -1.0e30f;
    if (r < NTOK && c < NTOK) v = table[rel[r * NTOK + c] * HEADS + h];
    biasM[idx] = v;
}

// ---------------- kernel: WMMA GEMM  C = A(bf16, MxK) * W^T(bf16, NxK) + bias --------
// block tile 128(M) x 64(N), 8 waves in 4x2, each wave 32x32, K step 32.
// A/B tiles staged by TDM (async DMA), double-buffered, paced by s_wait_tensorcnt.
template <bool OUT_BF16>
__global__ void __launch_bounds__(256) gemm_wmma_kernel(
    const unsigned short* __restrict__ A, int lda,
    const unsigned short* __restrict__ W, int ldw,
    const float* __restrict__ bias,
    unsigned short* __restrict__ outB,
    float* __restrict__ outF, int ldc,
    int K) {
    __shared__ unsigned short As[2][128 * 32];   // 2 x 8 KB
    __shared__ unsigned short Bs[2][64 * 32];    // 2 x 4 KB

    const int m0 = blockIdx.x * 128;
    const int n0 = blockIdx.y * 64;
    const int mTot = gridDim.x * 128;
    const int nTot = gridDim.y * 64;
    const int tid = threadIdx.x;
    const int wv = tid >> 5;
    const int l  = tid & 31;
    const int hf = (l >= 16) ? 1 : 0;
    const int lr = l & 15;
    const int wm = wv & 3;   // 4 waves along M (32 rows each)
    const int wn = wv >> 2;  // 2 waves along N (32 cols each)
    const int steps = K / 32;

    v8f acc[2][2] = {};

    if (wv == 0) {  // prologue: stage k-slab 0 into buffer 0
        tdm_load_2d_bf16(A + (size_t)m0 * lda, lds_offset_of(&As[0][0]),
                         (unsigned)K, (unsigned)(mTot - m0), 32u, 128u, (unsigned)lda);
        tdm_load_2d_bf16(W + (size_t)n0 * ldw, lds_offset_of(&Bs[0][0]),
                         (unsigned)K, (unsigned)(nTot - n0), 32u, 64u, (unsigned)ldw);
    }

    for (int s = 0; s < steps; ++s) {
        const int buf = s & 1;
        if (wv == 0) {
            if (s + 1 < steps) {   // stage next slab into the other buffer
                int k1 = (s + 1) * 32;
                tdm_load_2d_bf16(A + (size_t)m0 * lda + k1, lds_offset_of(&As[buf ^ 1][0]),
                                 (unsigned)(K - k1), (unsigned)(mTot - m0), 32u, 128u,
                                 (unsigned)lda);
                tdm_load_2d_bf16(W + (size_t)n0 * ldw + k1, lds_offset_of(&Bs[buf ^ 1][0]),
                                 (unsigned)(K - k1), (unsigned)(nTot - n0), 32u, 64u,
                                 (unsigned)ldw);
                __builtin_amdgcn_s_wait_tensorcnt(2);   // current slab's pair done
            } else {
                __builtin_amdgcn_s_wait_tensorcnt(0);
            }
        }
        __syncthreads();

        v16bf aF[2], bF[2];
#pragma unroll
        for (int i = 0; i < 2; ++i) {
            const unsigned short* ap = &As[buf][0] + (wm * 32 + i * 16 + lr) * 32 + hf * 16;
            aF[i] = mkfrag(*(const uint4*)ap, *(const uint4*)(ap + 8));
        }
#pragma unroll
        for (int j = 0; j < 2; ++j) {
            const unsigned short* bp = &Bs[buf][0] + (wn * 32 + j * 16 + lr) * 32 + hf * 16;
            bF[j] = mkfrag(*(const uint4*)bp, *(const uint4*)(bp + 8));
        }
#pragma unroll
        for (int i = 0; i < 2; ++i)
#pragma unroll
            for (int j = 0; j < 2; ++j)
                acc[i][j] = wmma_bf16(aF[i], bF[j], acc[i][j]);
        __syncthreads();
    }

    // epilogue: C layout VGPR v, lanes0-15 row=v, lanes16-31 row=v+8, col=lane&15
#pragma unroll
    for (int i = 0; i < 2; ++i)
#pragma unroll
        for (int j = 0; j < 2; ++j) {
            int col = n0 + wn * 32 + j * 16 + lr;
            float bv = bias[col];
#pragma unroll
            for (int v = 0; v < 8; ++v) {
                int row = m0 + wm * 32 + i * 16 + v + hf * 8;
                float val = acc[i][j][v] + bv;
                if (OUT_BF16) outB[(size_t)row * ldc + col] = f2bf(val);
                else          outF[(size_t)row * ldc + col] = val;
            }
        }
}

// ---------------- kernel: fused window attention per (b,h) ----------------
// qkv: ROWS x 1152 bf16 ; q col = h*32+d, k col = 384+h*32+d, v col = 768+h*32+d
// out: ROWS x 384 bf16, col = h*32+d
__global__ void __launch_bounds__(256) attn_kernel(const unsigned short* __restrict__ qkv,
                                                   const float* __restrict__ biasM,
                                                   unsigned short* __restrict__ out) {
    __shared__ unsigned short Ks[NPAD * 32];      // K row-major [n][d]  (22.0 KB)
    __shared__ unsigned short Vt[32 * NPAD];      // V transposed [d][n] (22.0 KB)
    __shared__ unsigned short Pst[8][16 * 32];    // per-wave P staging  ( 8.0 KB)

    const int bh = blockIdx.x;
    const int b = bh / HEADS, h = bh % HEADS;
    const int rowBase = b * NTOK;
    const int tid = threadIdx.x;
    const int wv = tid >> 5;

    // -------- K tile via TDM: 352 rows x 32 d from qkv, stride 1152 --------
    // Rows past the tensor end return zero; interior overshoot rows (next batch's
    // K, finite) are annihilated by the -1e30 padded bias in softmax.
    if (wv == 0) {
        tdm_load_2d_bf16(qkv + (size_t)rowBase * QKVC + DIM + h * HD,
                         lds_offset_of(&Ks[0]),
                         (unsigned)(QKVC - (DIM + h * HD)),
                         (unsigned)(ROWS - rowBase),
                         32u, (unsigned)NPAD, (unsigned)QKVC);
    }

    // -------- cooperative stage of V^T (transpose via LDS scatter) --------
    for (int n = tid; n < NPAD; n += 256) {
        if (n < NTOK) {
            const uint4* vs = (const uint4*)(qkv + (size_t)(rowBase + n) * QKVC + 2 * DIM + h * HD);
            uint4 v0 = vs[0], v1 = vs[1], v2 = vs[2], v3 = vs[3];
            unsigned int w[16] = {v0.x, v0.y, v0.z, v0.w, v1.x, v1.y, v1.z, v1.w,
                                  v2.x, v2.y, v2.z, v2.w, v3.x, v3.y, v3.z, v3.w};
#pragma unroll
            for (int d2 = 0; d2 < 16; ++d2) {
                Vt[(2 * d2) * NPAD + n]     = (unsigned short)(w[d2] & 0xFFFFu);
                Vt[(2 * d2 + 1) * NPAD + n] = (unsigned short)(w[d2] >> 16);
            }
        } else {
#pragma unroll
            for (int d = 0; d < 32; ++d) Vt[d * NPAD + n] = 0;
        }
    }
    if (wv == 0) __builtin_amdgcn_s_wait_tensorcnt(0);
    __syncthreads();

    const int l  = tid & 31;
    const int hf = (l >= 16) ? 1 : 0;
    const int lr = l & 15;
    unsigned short* stg = &Pst[wv][0];
    const float scale = 0.17677669529663687f;  // 32^-0.5

    for (int mb = wv; mb < 22; mb += 8) {   // 22 tiles of 16 rows cover 343
        // ---- Q A-fragment straight from global (K dim = 32 = one WMMA step) ----
        int r0 = mb * 16 + lr; if (r0 > NTOK - 1) r0 = NTOK - 1;
        const unsigned short* qp = qkv + (size_t)(rowBase + r0) * QKVC + h * HD + hf * 16;
        v16bf aQ = mkfrag(*(const uint4*)qp, *(const uint4*)(qp + 8));

        // ---- S = (Q K^T) : one WMMA per 16-wide tile, 22 tiles ----
        v8f S[22];
#pragma unroll
        for (int t = 0; t < 22; ++t) {
            const uint4* kp = (const uint4*)(Ks + (t * 16 + lr) * 32 + hf * 16);
            v16bf bK = mkfrag(kp[0], kp[1]);
            v8f z = {};
            S[t] = wmma_bf16(aQ, bK, z);
        }

        // ---- scale + bias + row max ----
        float mx[8];
#pragma unroll
        for (int j = 0; j < 8; ++j) mx[j] = -3.0e38f;
        const float* bbase = biasM + ((size_t)h * NPAD + mb * 16 + hf * 8) * NPAD;
#pragma unroll
        for (int t = 0; t < 22; ++t) {
            const float* bp = bbase + t * 16 + lr;
#pragma unroll
            for (int j = 0; j < 8; ++j) {
                float v = fmaf(S[t][j], scale, bp[(size_t)j * NPAD]);
                S[t][j] = v;
                mx[j] = fmaxf(mx[j], v);
            }
        }
#pragma unroll
        for (int j = 0; j < 8; ++j) {
            float m = mx[j];
            m = fmaxf(m, __shfl_xor(m, 1, 32));
            m = fmaxf(m, __shfl_xor(m, 2, 32));
            m = fmaxf(m, __shfl_xor(m, 4, 32));
            m = fmaxf(m, __shfl_xor(m, 8, 32));   // stays within 16-lane column group
            mx[j] = m;
        }

        // ---- exp + row sum ----
        float sm[8];
#pragma unroll
        for (int j = 0; j < 8; ++j) sm[j] = 0.0f;
#pragma unroll
        for (int t = 0; t < 22; ++t)
#pragma unroll
            for (int j = 0; j < 8; ++j) {
                float e = __expf(S[t][j] - mx[j]);
                S[t][j] = e;
                sm[j] += e;
            }
        float inv[8];
#pragma unroll
        for (int j = 0; j < 8; ++j) {
            float s = sm[j];
            s += __shfl_xor(s, 1, 32);
            s += __shfl_xor(s, 2, 32);
            s += __shfl_xor(s, 4, 32);
            s += __shfl_xor(s, 8, 32);
            inv[j] = 1.0f / s;
        }

        // ---- O = P V : re-layout P (C-layout -> A-layout) via 1KB LDS staging ----
        v8f O0 = {}, O1 = {};
        for (int kk = 0; kk < 11; ++kk) {   // 352 / 32
#pragma unroll
            for (int tt = 0; tt < 2; ++tt) {
                int t = 2 * kk + tt;
#pragma unroll
                for (int j = 0; j < 8; ++j) {
                    int rr = j + hf * 8;
                    stg[rr * 32 + tt * 16 + lr] = f2bf(S[t][j] * inv[j]);
                }
            }
            // A fragment of P from staging (same-wave LDS ordering via DScnt)
            const unsigned short* pp = stg + lr * 32 + hf * 16;
            v16bf aP = mkfrag(*(const uint4*)pp, *(const uint4*)(pp + 8));
            // B fragments from Vt: column d, 16 consecutive n
            int nOff = kk * 32 + hf * 16;
            {
                const uint4* vp = (const uint4*)(Vt + (size_t)lr * NPAD + nOff);
                O0 = wmma_bf16(aP, mkfrag(vp[0], vp[1]), O0);
            }
            {
                const uint4* vp = (const uint4*)(Vt + (size_t)(16 + lr) * NPAD + nOff);
                O1 = wmma_bf16(aP, mkfrag(vp[0], vp[1]), O1);
            }
        }

        // ---- store O (bf16, col = h*32 + d) ----
#pragma unroll
        for (int j = 0; j < 8; ++j) {
            int row = mb * 16 + j + hf * 8;
            if (row < NTOK) {
                unsigned short* op = out + (size_t)(rowBase + row) * DIM + h * HD;
                op[lr]      = f2bf(O0[j]);
                op[16 + lr] = f2bf(O1[j]);
            }
        }
    }
}

// ---------------- host-side orchestration ----------------
extern "C" void kernel_launch(void* const* d_in, const int* in_sizes, int n_in,
                              void* d_out, int out_size, void* d_ws, size_t ws_size,
                              hipStream_t stream) {
    const float* x      = (const float*)d_in[0];  // (128,343,384)
    const float* qkv_w  = (const float*)d_in[1];  // (1152,384)
    const float* qkv_b  = (const float*)d_in[2];  // (1152,)
    const float* proj_w = (const float*)d_in[3];  // (384,384)
    const float* proj_b = (const float*)d_in[4];  // (384,)
    const float* table  = (const float*)d_in[5];  // (2197,12)
    const int*   rel    = (const int*)d_in[6];    // (343,343)
    float* out = (float*)d_out;                   // (128,343,384)

    char* ws = (char*)d_ws;
    size_t off = 0;
    auto carve = [&](size_t bytes) {
        void* p = ws + off;
        off = (off + bytes + 255) & ~(size_t)255;
        return p;
    };
    unsigned short* xbf    = (unsigned short*)carve((size_t)ROWS * DIM * 2);
    unsigned short* qkvwbf = (unsigned short*)carve((size_t)QKVC * DIM * 2);
    unsigned short* pwbf   = (unsigned short*)carve((size_t)DIM * DIM * 2);
    unsigned short* qkvbf  = (unsigned short*)carve((size_t)ROWS * QKVC * 2);
    unsigned short* attnbf = (unsigned short*)carve((size_t)ROWS * DIM * 2);
    float*          biasM  = (float*)carve((size_t)HEADS * NPAD * NPAD * 4);

    // 1) convert inputs to bf16
    {
        int n = ROWS * DIM;
        cvt_bf16_kernel<<<(n + 255) / 256, 256, 0, stream>>>(x, xbf, n);
        n = QKVC * DIM;
        cvt_bf16_kernel<<<(n + 255) / 256, 256, 0, stream>>>(qkv_w, qkvwbf, n);
        n = DIM * DIM;
        cvt_bf16_kernel<<<(n + 255) / 256, 256, 0, stream>>>(proj_w, pwbf, n);
    }
    // 2) dense padded bias
    {
        int n = HEADS * NPAD * NPAD;
        build_bias_kernel<<<(n + 255) / 256, 256, 0, stream>>>(table, rel, biasM);
    }
    // 3) QKV GEMM: (43904x384) * (1152x384)^T -> bf16 (43904x1152)
    {
        dim3 grid(ROWS / 128, QKVC / 64);
        gemm_wmma_kernel<true><<<grid, 256, 0, stream>>>(
            xbf, DIM, qkvwbf, DIM, qkv_b, qkvbf, nullptr, QKVC, DIM);
    }
    // 4) fused attention, one workgroup per (batch, head)
    {
        attn_kernel<<<BATCH * HEADS, 256, 0, stream>>>(qkvbf, biasM, attnbf);
    }
    // 5) projection GEMM: (43904x384) * (384x384)^T + bias -> f32 out
    {
        dim3 grid(ROWS / 128, DIM / 64);
        gemm_wmma_kernel<false><<<grid, 256, 0, stream>>>(
            attnbf, DIM, pwbf, DIM, proj_b, nullptr, out, DIM, DIM);
    }
}
